// ESN_61641370632828
// MI455X (gfx1250) — matrix operand order repaired and checked by reading, not compiled, not run on hardware
//
#include <hip/hip_runtime.h>
#include <hip/hip_bf16.h>
#include <math.h>

// ---------------------------------------------------------------------------
// ESN reservoir pipeline for MI455X (gfx1250, wave32, WMMA, async-LDS).
//   drive = u @ W_in^T                    -> bf16 WMMA GEMM (K=64)
//   x_{t+1} = tanh(W_res x_t + drive_t)   -> persistent kernel, grid barrier,
//                                            async global->LDS x staging
//   A = Xw^T Xw + lambda I                -> bf16 WMMA GEMM, LDS b128 frags
//   solve A w = Xw^T y                    -> fixed-iteration CG (grid-wide)
//   out = Xw w                            -> matvec epilogue
// ---------------------------------------------------------------------------

#define NH     2048
#define DIN    64
#define TSTEP  4096
#define WASH   200
#define TW     (TSTEP - WASH)   // 3896
#define RIDGE  1e-6f
#define CG_ITERS 64

typedef __attribute__((ext_vector_type(16))) __bf16 v16bf;
typedef __attribute__((ext_vector_type(8)))  float  v8f;
typedef __attribute__((ext_vector_type(4)))  int    v4i;

#define AS_G __attribute__((address_space(1)))
#define AS_L __attribute__((address_space(3)))

// ----- CDNA5 async global->LDS copy (guarded; falls back to volatile loads) -
#if defined(__has_builtin)
#if __has_builtin(__builtin_amdgcn_global_load_async_to_lds_b128)
#define HAVE_ASYNC_COPY 1
#else
#define HAVE_ASYNC_COPY 0
#endif
#if __has_builtin(__builtin_amdgcn_s_wait_asynccnt)
#define WAIT_ASYNC() __builtin_amdgcn_s_wait_asynccnt(0)
#else
#define WAIT_ASYNC() asm volatile("s_wait_asynccnt 0x0" ::: "memory")
#endif
#else
#define HAVE_ASYNC_COPY 0
#define WAIT_ASYNC()
#endif

// Copy n f32 (n multiple of 1024) from global to LDS with 256 threads.
__device__ __forceinline__ void lds_copy_f32(float* dstLds,
                                             const float* src, int tid,
                                             int n) {
#if HAVE_ASYNC_COPY
  for (int i = tid * 4; i < n; i += 256 * 4)
    __builtin_amdgcn_global_load_async_to_lds_b128(
        (AS_G v4i*)(src + i), (AS_L v4i*)(dstLds + i), 0, 0);
  WAIT_ASYNC();
#else
  volatile const float* vs = src;
  for (int i = tid; i < n; i += 256) dstLds[i] = vs[i];
#endif
}

// ----- workspace layout (bytes) --------------------------------------------
static constexpr size_t OFF_DRIVE = 0;                                   // f32 T*NH   (32 MB)
static constexpr size_t OFF_XBF   = OFF_DRIVE + (size_t)TSTEP * NH * 4;  // bf16 T*NH  (16 MB)
static constexpr size_t OFF_A     = OFF_XBF   + (size_t)TSTEP * NH * 2;  // f32 NH*NH  (16 MB)
static constexpr size_t OFF_UBF   = OFF_A     + (size_t)NH * NH * 4;     // bf16 T*DIN
static constexpr size_t OFF_WIBF  = OFF_UBF   + (size_t)TSTEP * DIN * 2; // bf16 NH*DIN
static constexpr size_t OFF_XPING = OFF_WIBF  + (size_t)NH * DIN * 2;    // f32 2*NH
static constexpr size_t OFF_B     = OFF_XPING + (size_t)2 * NH * 4;      // f32 NH
static constexpr size_t OFF_W     = OFF_B     + (size_t)NH * 4;
static constexpr size_t OFF_R     = OFF_W     + (size_t)NH * 4;
static constexpr size_t OFF_P     = OFF_R     + (size_t)NH * 4;
static constexpr size_t OFF_SLOTS = OFF_P     + (size_t)NH * 4;          // pAp[it], rr[it]
static constexpr size_t OFF_BAR   = OFF_SLOTS + 4096;                    // barrier cnt+gen

// ----- helpers -------------------------------------------------------------
__device__ __forceinline__ float bf2f(unsigned short u) {
  return __builtin_bit_cast(float, (unsigned)u << 16);
}
__device__ __forceinline__ unsigned short f2bf(float f) {
  unsigned u = __builtin_bit_cast(unsigned, f);
  u += 0x7fffu + ((u >> 16) & 1u);        // round-to-nearest-even
  return (unsigned short)(u >> 16);
}
__device__ __forceinline__ v8f wmma_bf16(v16bf a, v16bf b, v8f c) {
  return __builtin_amdgcn_wmma_f32_16x16x32_bf16(false, a, false, b,
                                                 (short)0, c, false, false);
}

// Grid-wide sense-reversing barrier (state in d_ws, zeroed each launch).
__device__ __forceinline__ void grid_barrier(unsigned* cnt, unsigned* gen,
                                             unsigned nblocks) {
  __syncthreads();
  if (threadIdx.x == 0) {
    __threadfence();
    unsigned g = __atomic_load_n(gen, __ATOMIC_ACQUIRE);
    if (atomicAdd(cnt, 1u) == nblocks - 1u) {
      __atomic_store_n(cnt, 0u, __ATOMIC_RELAXED);
      __atomic_fetch_add(gen, 1u, __ATOMIC_RELEASE);
    } else {
      while (__atomic_load_n(gen, __ATOMIC_ACQUIRE) == g)
        __builtin_amdgcn_s_sleep(1);
    }
  }
  __syncthreads();
}

// ----- f32 -> bf16 conversion ----------------------------------------------
__global__ void cvt_bf16_k(const float* __restrict__ src,
                           unsigned short* __restrict__ dst, int n) {
  for (int i = blockIdx.x * blockDim.x + threadIdx.x; i < n;
       i += gridDim.x * blockDim.x)
    dst[i] = f2bf(src[i]);
}

// ----- drive = u @ W_in^T  (M=4096, N=2048, K=64), bf16 WMMA ---------------
__global__ void __launch_bounds__(256) drive_wmma_k(
    const unsigned short* __restrict__ ubf,
    const unsigned short* __restrict__ wibf,
    float* __restrict__ drive) {
  int wave = blockIdx.x * 8 + (threadIdx.x >> 5);   // 8192 waves total
  int lane = threadIdx.x & 31, hf = lane >> 4, l = lane & 15;
  int ti = wave >> 5;          // 0..255  (t strip)
  int nj = wave & 31;          // 0..31   (n strip of 64)
  int t0 = ti * 16, n0 = nj * 64;

  union Frag { uint4 q[2]; unsigned u[8]; v16bf v; };
  Frag a0, a1;
  const unsigned short* urow = ubf + (size_t)(t0 + l) * DIN;
  // A 16x32 layout: dwords 0..3 = K {0..7}+8*hf, dwords 4..7 = K {16..23}+8*hf
  a0.q[0] = *(const uint4*)(urow + hf * 8);
  a0.q[1] = *(const uint4*)(urow + 16 + hf * 8);
  a1.q[0] = *(const uint4*)(urow + 32 + hf * 8);
  a1.q[1] = *(const uint4*)(urow + 48 + hf * 8);
#pragma unroll
  for (int j = 0; j < 4; j++) {
    Frag b0, b1;
    const unsigned short* wrow = wibf + (size_t)(n0 + j * 16 + l) * DIN;
    // B 32x16 layout: dwords 0..3 = K {0..7}+8*hf, dwords 4..7 = K {16..23}+8*hf
    b0.q[0] = *(const uint4*)(wrow + hf * 8);
    b0.q[1] = *(const uint4*)(wrow + 16 + hf * 8);
    b1.q[0] = *(const uint4*)(wrow + 32 + hf * 8);
    b1.q[1] = *(const uint4*)(wrow + 48 + hf * 8);
    v8f c = {};
    c = wmma_bf16(a0.v, b0.v, c);
    c = wmma_bf16(a1.v, b1.v, c);
#pragma unroll
    for (int r = 0; r < 8; r++)
      drive[(size_t)(t0 + hf * 8 + r) * NH + n0 + j * 16 + l] = c[r];
  }
}

// ----- sequential reservoir scan (persistent, grid barrier per step) -------
// 128 blocks x 256 threads = 32768 lanes = 2048 rows x 16 lanes/row.
// W_res (16 MB fp32) is L2-resident; x staged into LDS via async copy.
__global__ void __launch_bounds__(256, 1) scan_k(
    const float* __restrict__ Wres, const float* __restrict__ drive,
    float* xping, unsigned short* __restrict__ Xbf, unsigned* bar) {
  __shared__ __align__(16) float xs[NH];
  int tid = threadIdx.x;
  int g = blockIdx.x * 256 + tid;
  int row = g >> 4, sub = g & 15;
  unsigned* cnt = bar;
  unsigned* gen = bar + 1;
  const float4* wrow =
      (const float4*)(Wres + (size_t)row * NH + sub * 128);

  for (int t = 0; t < TSTEP; t++) {
    lds_copy_f32(xs, xping + (t & 1) * NH, tid, NH);
    __syncthreads();
    const float4* xv = (const float4*)(xs + sub * 128);
    float acc = 0.f;
#pragma unroll 8
    for (int c = 0; c < 32; c++) {
      float4 w = wrow[c];
      float4 x = xv[c];
      acc += w.x * x.x + w.y * x.y + w.z * x.z + w.w * x.w;
    }
    acc += __shfl_xor(acc, 1);
    acc += __shfl_xor(acc, 2);
    acc += __shfl_xor(acc, 4);
    acc += __shfl_xor(acc, 8);
    if (sub == 0) {
      float v = tanhf(acc + drive[(size_t)t * NH + row]);
      xping[((t + 1) & 1) * NH + row] = v;
      Xbf[(size_t)t * NH + row] = f2bf(v);
      if (t + 1 < TSTEP)          // pull next step's drive row toward L0
        __builtin_prefetch(&drive[(size_t)(t + 1) * NH + row], 0, 1);
    }
    grid_barrier(cnt, gen, gridDim.x);   // includes entry __syncthreads
  }
}

// ----- A = Xw^T Xw + lambda I  (bf16 WMMA, K staged transposed in LDS) -----
// WG tile 64(i) x 128(j), 8 waves of 2x2 16x16 accumulators, K-step 32.
// LDS tiles stored [n][k] with k contiguous, row stride 40 ushorts (80 B,
// 16 B aligned, bank stride 20 -> conflict-free) so each WMMA fragment is
// exactly two aligned ds_load_b128.
#define TIS 40
__global__ void __launch_bounds__(256) gram_wmma_k(
    const unsigned short* __restrict__ Xbf, float* __restrict__ Amat) {
  __shared__ __align__(16) unsigned short tI[64 * TIS];
  __shared__ __align__(16) unsigned short tJ[128 * TIS];
  int tid = threadIdx.x, lane = tid & 31, hf = lane >> 4, l = lane & 15;
  int w = tid >> 5, wi = w >> 2, wj = w & 3;
  int i0 = (blockIdx.x >> 4) * 64, j0 = (blockIdx.x & 15) * 128;

  v8f c[2][2] = {};
  for (int ks = 0; ks < 122; ks++) {                    // ceil(3896/32)
    int kk = WASH + ks * 32;
    __syncthreads();
    {   // stage tI: 32 t x 64 n, transposed into [n][k]
      int tl = tid >> 3, cc = tid & 7;
      int t = kk + tl;
      uint4 vv = make_uint4(0, 0, 0, 0);
      if (t < TSTEP)
        vv = *(const uint4*)(Xbf + (size_t)t * NH + i0 + cc * 8);
      const unsigned uu[4] = {vv.x, vv.y, vv.z, vv.w};
#pragma unroll
      for (int q = 0; q < 4; q++) {
        tI[(cc * 8 + 2 * q + 0) * TIS + tl] = (unsigned short)(uu[q] & 0xffffu);
        tI[(cc * 8 + 2 * q + 1) * TIS + tl] = (unsigned short)(uu[q] >> 16);
      }
    }
#pragma unroll
    for (int pass = 0; pass < 2; pass++) {              // stage tJ: 32 t x 128 n
      int tl = (tid >> 4) + pass * 16, cc = tid & 15;
      int t = kk + tl;
      uint4 vv = make_uint4(0, 0, 0, 0);
      if (t < TSTEP)
        vv = *(const uint4*)(Xbf + (size_t)t * NH + j0 + cc * 8);
      const unsigned uu[4] = {vv.x, vv.y, vv.z, vv.w};
#pragma unroll
      for (int q = 0; q < 4; q++) {
        tJ[(cc * 8 + 2 * q + 0) * TIS + tl] = (unsigned short)(uu[q] & 0xffffu);
        tJ[(cc * 8 + 2 * q + 1) * TIS + tl] = (unsigned short)(uu[q] >> 16);
      }
    }
    __syncthreads();

    union Frag { uint4 q[2]; v16bf v; };
    Frag a[2], b[2];
#pragma unroll
    for (int s = 0; s < 2; s++) {
      int m = wi * 32 + s * 16 + l;
      int n = wj * 32 + s * 16 + l;
      // A 16x32: dwords 0..3 at K base 8*hf, dwords 4..7 at K base 16+8*hf
      a[s].q[0] = *(const uint4*)&tI[m * TIS + hf * 8];
      a[s].q[1] = *(const uint4*)&tI[m * TIS + 16 + hf * 8];
      // B 32x16: dwords 0..3 at K base 8*hf, dwords 4..7 at K base 16+8*hf
      b[s].q[0] = *(const uint4*)&tJ[n * TIS + hf * 8];
      b[s].q[1] = *(const uint4*)&tJ[n * TIS + 16 + hf * 8];
    }
#pragma unroll
    for (int s = 0; s < 2; s++)
#pragma unroll
      for (int t2 = 0; t2 < 2; t2++)
        c[s][t2] = wmma_bf16(a[s].v, b[t2].v, c[s][t2]);
  }
#pragma unroll
  for (int s = 0; s < 2; s++)
#pragma unroll
    for (int t2 = 0; t2 < 2; t2++)
#pragma unroll
      for (int r = 0; r < 8; r++) {
        int gr = i0 + wi * 32 + s * 16 + hf * 8 + r;
        int gc = j0 + wj * 32 + t2 * 16 + l;
        float val = c[s][t2][r];
        if (gr == gc) val += RIDGE;
        Amat[(size_t)gr * NH + gc] = val;
      }
}

// ----- b = Xw^T y -----------------------------------------------------------
__global__ void bvec_k(const unsigned short* __restrict__ Xbf,
                       const float* __restrict__ y, float* __restrict__ b) {
  int n = blockIdx.x * 256 + threadIdx.x;
  float acc = 0.f;
  for (int t = WASH; t < TSTEP; t++)
    acc += bf2f(Xbf[(size_t)t * NH + n]) * y[t];
  b[n] = acc;
}

// ----- CG solve A w = b (grid-wide, per-iteration slots, no counter reset) --
// 64 blocks x 256 = 16384 lanes = 2048 rows x 8 lanes/row.
__global__ void __launch_bounds__(256, 1) cg_k(
    const float* __restrict__ A, const float* __restrict__ b, float* w,
    float* r, float* p, float* slots, unsigned* bar) {
  __shared__ __align__(16) float ps[NH];
  __shared__ float sred;
  int tid = threadIdx.x;
  int g = blockIdx.x * 256 + tid;
  int row = g >> 3, sub = g & 7;
  bool owner = (sub == 0);
  unsigned* cnt = bar;
  unsigned* gen = bar + 1;
  volatile float* vs = slots;     // [0..CG): pAp, [CG..2*CG]: rr

  if (owner) {
    float bv = b[row];
    p[row] = bv;
    r[row] = bv;
    w[row] = 0.f;
    atomicAdd(&slots[CG_ITERS], bv * bv);   // rr[0]
  }
  grid_barrier(cnt, gen, gridDim.x);

  const float4* arow = (const float4*)(A + (size_t)row * NH + sub * 256);
  for (int it = 0; it < CG_ITERS; it++) {
    lds_copy_f32(ps, p, tid, NH);
    __syncthreads();
    const float4* xv = (const float4*)(ps + sub * 256);
    float acc = 0.f;
#pragma unroll 8
    for (int c = 0; c < 64; c++) {
      float4 av = arow[c];
      float4 xw = xv[c];
      acc += av.x * xw.x + av.y * xw.y + av.z * xw.z + av.w * xw.w;
    }
    acc += __shfl_xor(acc, 1);
    acc += __shfl_xor(acc, 2);
    acc += __shfl_xor(acc, 4);              // Ap[row] in sub==0 lanes
    __syncthreads();
    if (tid == 0) sred = 0.f;
    __syncthreads();
    if (owner) atomicAdd(&sred, ps[row] * acc);
    __syncthreads();
    if (tid == 0) atomicAdd(&slots[it], sred);
    grid_barrier(cnt, gen, gridDim.x);

    float rrold = vs[CG_ITERS + it];
    float alpha = rrold / (vs[it] + 1e-30f);
    float rv = 0.f;
    if (owner) {
      w[row] += alpha * ps[row];
      rv = r[row] - alpha * acc;
      r[row] = rv;
    }
    __syncthreads();
    if (tid == 0) sred = 0.f;
    __syncthreads();
    if (owner) atomicAdd(&sred, rv * rv);
    __syncthreads();
    if (tid == 0) atomicAdd(&slots[CG_ITERS + it + 1], sred);
    grid_barrier(cnt, gen, gridDim.x);

    float beta = vs[CG_ITERS + it + 1] / (rrold + 1e-30f);
    if (owner) p[row] = r[row] + beta * p[row];
    grid_barrier(cnt, gen, gridDim.x);
  }
}

// ----- out = Xw @ w ---------------------------------------------------------
__global__ void out_k(const unsigned short* __restrict__ Xbf,
                      const float* __restrict__ w, float* __restrict__ out) {
  __shared__ __align__(16) float wl[NH];
  int tid = threadIdx.x;
  lds_copy_f32(wl, w, tid, NH);
  __syncthreads();
  int idx = blockIdx.x * 256 + tid;
  if (idx < TW) {
    const unsigned* xrow = (const unsigned*)(Xbf + (size_t)(idx + WASH) * NH);
    float acc = 0.f;
#pragma unroll 4
    for (int c = 0; c < NH / 2; c++) {
      unsigned uu = xrow[c];
      acc += bf2f((unsigned short)(uu & 0xffffu)) * wl[2 * c] +
             bf2f((unsigned short)(uu >> 16)) * wl[2 * c + 1];
    }
    out[idx] = acc;
  }
}

// ---------------------------------------------------------------------------
extern "C" void kernel_launch(void* const* d_in, const int* in_sizes, int n_in,
                              void* d_out, int out_size, void* d_ws,
                              size_t ws_size, hipStream_t stream) {
  (void)in_sizes; (void)n_in; (void)out_size; (void)ws_size;
  const float* u    = (const float*)d_in[0];   // (4096, 64)
  const float* y    = (const float*)d_in[1];   // (4096, 1)
  const float* Win  = (const float*)d_in[2];   // (2048, 64)
  const float* Wres = (const float*)d_in[3];   // (2048, 2048)

  char* base = (char*)d_ws;
  float*          drive = (float*)(base + OFF_DRIVE);
  unsigned short* Xbf   = (unsigned short*)(base + OFF_XBF);
  float*          Amat  = (float*)(base + OFF_A);
  unsigned short* ubf   = (unsigned short*)(base + OFF_UBF);
  unsigned short* wibf  = (unsigned short*)(base + OFF_WIBF);
  float*          xping = (float*)(base + OFF_XPING);
  float*          bvec  = (float*)(base + OFF_B);
  float*          wsol  = (float*)(base + OFF_W);
  float*          rvec  = (float*)(base + OFF_R);
  float*          pvec  = (float*)(base + OFF_P);
  float*          slots = (float*)(base + OFF_SLOTS);
  unsigned*       bar   = (unsigned*)(base + OFF_BAR);

  // zero: x0 ping-pong, CG slot accumulators, barrier state (capture-safe)
  (void)hipMemsetAsync(base + OFF_XPING, 0, 2 * NH * 4, stream);
  (void)hipMemsetAsync(base + OFF_SLOTS, 0, 4096 + 256, stream);

  cvt_bf16_k<<<1024, 256, 0, stream>>>(u, ubf, TSTEP * DIN);
  cvt_bf16_k<<<512, 256, 0, stream>>>(Win, wibf, NH * DIN);
  drive_wmma_k<<<1024, 256, 0, stream>>>(ubf, wibf, drive);
  scan_k<<<128, 256, 0, stream>>>(Wres, drive, xping, Xbf, bar);
  gram_wmma_k<<<512, 256, 0, stream>>>(Xbf, Amat);
  bvec_k<<<NH / 256, 256, 0, stream>>>(Xbf, y, bvec);
  cg_k<<<64, 256, 0, stream>>>(Amat, bvec, wsol, rvec, pvec, slots, bar);
  out_k<<<(TW + 255) / 256, 256, 0, stream>>>(Xbf, wsol, (float*)d_out);
}
